// APostModel_22874995818938
// MI455X (gfx1250) — compile-verified, matrix operand order checked by reading, not run on hardware
//
#include <hip/hip_runtime.h>
#include <hip/hip_bf16.h>
#include <math.h>

// ---------------------------------------------------------------------------
// CDNA5 (gfx1250, wave32) implementation of YOLO-style detection post-process.
//   K1: max-logit key per anchor (sigmoid is monotone -> rank on logits)
//   K2: exact top-1000 per (image,level) via 4-pass radix select on uint keys
//   K3: decode selected anchors; DFL via V_WMMA_F32_16X16X32_F16 (split-f16
//       hi/lo double-WMMA accumulation for ~f32 accuracy)
//   K4: per-(image,class) greedy NMS, fully LDS-resident (scores+boxes)
//   K5: per-image global top-100 + thresholding/assembly
// ---------------------------------------------------------------------------

typedef __attribute__((ext_vector_type(16))) _Float16 v16h;
typedef __attribute__((ext_vector_type(8)))  float    v8f;

#define NINF (-__builtin_huge_valf())

static constexpr int B_   = 8;
static constexpr int C_   = 80;
static constexpr int NPRE = 1000;   // anchors kept per level
static constexpr int NTOT = 3000;   // 3 levels * NPRE
static constexpr int K_   = 100;    // top_k

// ---------------- Kernel 1: per-anchor max logit -> sortable uint key ------
__global__ void k_maxkey(const float* __restrict__ cls,
                         unsigned* __restrict__ key, int total) {
  int t = blockIdx.x * 256 + threadIdx.x;
  if (t >= total) return;
  const float* p = cls + (size_t)t * C_;
  float m = p[0];
  #pragma unroll 4
  for (int c = 1; c < C_; ++c) m = fmaxf(m, p[c]);
  unsigned u = __float_as_uint(m);
  u ^= ((unsigned)((int)u >> 31)) | 0x80000000u;   // monotone float->uint
  key[t] = u;
}

// ---------------- Kernel 2: radix-select top-NPRE per (image,level) --------
__global__ void k_select(const unsigned* __restrict__ maxkey,
                         int* __restrict__ sel_idx) {
  __shared__ unsigned hist[256];
  __shared__ unsigned sh_prefix;
  __shared__ int sh_k, cntEq, cntOut;
  const int lvl = blockIdx.x, b = blockIdx.y, tid = threadIdx.x;
  int N, loff;
  if (lvl == 0)      { N = 16384; loff = 0; }
  else if (lvl == 1) { N = 4096;  loff = 8 * 16384; }
  else               { N = 1024;  loff = 8 * 16384 + 8 * 4096; }
  const unsigned* keys = maxkey + loff + (size_t)b * N;

  if (tid == 0) { sh_prefix = 0u; sh_k = NPRE; }
  for (int p = 3; p >= 0; --p) {
    hist[tid] = 0u;
    __syncthreads();
    unsigned pref = sh_prefix;
    for (int n = tid; n < N; n += 256) {
      unsigned key = keys[n];
      bool match = (p == 3) || ((key >> ((p + 1) * 8)) == (pref >> ((p + 1) * 8)));
      if (match) atomicAdd(&hist[(key >> (p * 8)) & 0xFF], 1u);
    }
    __syncthreads();
    if (tid == 0) {
      int k = sh_k, cum = 0, selb = 0;
      for (int bb = 255; bb >= 0; --bb) {
        int h = (int)hist[bb];
        if (cum + h >= k) { selb = bb; sh_k = k - cum; break; }
        cum += h;
      }
      sh_prefix = pref | ((unsigned)selb << (p * 8));
    }
    __syncthreads();
  }
  const unsigned T = sh_prefix;
  const int krem = sh_k;                    // # of ==T values still needed
  if (tid == 0) { cntEq = 0; cntOut = 0; }
  __syncthreads();
  int* out = sel_idx + ((b * 3 + lvl) * NPRE);
  for (int n = tid; n < N; n += 256) {
    unsigned key = keys[n];
    if (key > T) {
      int s = atomicAdd(&cntOut, 1);
      out[s] = n;
    } else if (key == T) {
      int e = atomicAdd(&cntEq, 1);
      if (e < krem) { int s = atomicAdd(&cntOut, 1); out[s] = n; }
    }
  }
}

// ---------------- Kernel 3: decode (DFL via WMMA) + sigmoid scores ---------
// One wave (32 lanes) per tile of 16 anchors. Wave-level WMMA:
//   A[m,k] (16x32 f16): exp(bp - per-side max) for anchor m, k = side*8+bin
//   B[k,n] (32x16 f16): n=0..3 -> bin weights of side n (numerator)
//                       n=4..7 -> ones for side n-4     (denominator)
//   D[m,n] (16x16 f32): dist[m,s] = D[m,s]/D[m,s+4] * stride
// Split f16 hi/lo -> two accumulating WMMAs restores ~f32 accuracy.
__global__ void k_decode(const float* __restrict__ cls0, const float* __restrict__ bp0,
                         const float* __restrict__ cls1, const float* __restrict__ bp1,
                         const float* __restrict__ cls2, const float* __restrict__ bp2,
                         const int* __restrict__ sel_idx,
                         float* __restrict__ boxes,
                         float* __restrict__ scoresT) {
  __shared__ float ldsDist[16][4];
  const int lane = threadIdx.x;       // 0..31
  const int tile = blockIdx.x;        // 0..62
  const int lvl  = blockIdx.y;        // 0..2
  const int b    = blockIdx.z;        // 0..7

  const float* cls; const float* bp; int HW, W; float stride;
  if (lvl == 0)      { cls = cls0; bp = bp0; HW = 16384; W = 128; stride = 8.f;  }
  else if (lvl == 1) { cls = cls1; bp = bp1; HW = 4096;  W = 64;  stride = 16.f; }
  else               { cls = cls2; bp = bp2; HW = 1024;  W = 32;  stride = 32.f; }

  const int m  = lane & 15;           // A row / B column held by this lane
  const int hi = lane >> 4;           // half-wave selector
  const int j  = tile * 16 + m;       // anchor slot within the 1000
  const int jc = (j < NPRE) ? j : (NPRE - 1);
  const int sel = sel_idx[(b * 3 + lvl) * NPRE + jc];

  // ---- A matrix: exp(x - max) per side, split into f16 hi + lo ----
  const int khalf = hi ? 8 : 0;       // this lane holds K = khalf+{0..7,16..23}
  const float* bpp = bp + ((size_t)(b * HW + sel)) * 32;
  float x0[8], x1[8];
  #pragma unroll
  for (int e = 0; e < 8; ++e) { x0[e] = bpp[khalf + e]; x1[e] = bpp[khalf + 16 + e]; }
  float mx0 = x0[0], mx1 = x1[0];
  #pragma unroll
  for (int e = 1; e < 8; ++e) { mx0 = fmaxf(mx0, x0[e]); mx1 = fmaxf(mx1, x1[e]); }
  v16h ah, al;
  #pragma unroll
  for (int e = 0; e < 8; ++e) {
    float p0 = expf(x0[e] - mx0);
    float p1 = expf(x1[e] - mx1);
    _Float16 h0 = (_Float16)p0, h1 = (_Float16)p1;
    ah[e]     = h0; al[e]     = (_Float16)(p0 - (float)h0);
    ah[e + 8] = h1; al[e + 8] = (_Float16)(p1 - (float)h1);
  }

  // ---- constant B matrix (block-diagonal: weights | ones) ----
  v16h bm;
  const int n = m;                    // column index for this lane
  #pragma unroll
  for (int e = 0; e < 16; ++e) {
    int k = (e < 8) ? (khalf + e) : (khalf + 8 + e);   // K index of element e
    float val = 0.f;
    if (n < 8 && ((k >> 3) == (n & 3))) val = (n < 4) ? (float)(k & 7) : 1.0f;
    bm[e] = (_Float16)val;
  }

  // ---- two accumulating WMMAs (hi + lo) ----
  v8f acc = {};
  acc = __builtin_amdgcn_wmma_f32_16x16x32_f16(false, ah, false, bm,
                                               (short)0, acc, false, false);
  acc = __builtin_amdgcn_wmma_f32_16x16x32_f16(false, al, false, bm,
                                               (short)0, acc, false, false);

  // lanes with column n<4 pair numerator with denominator from lane+4
  #pragma unroll
  for (int r = 0; r < 8; ++r) {
    float num = acc[r];
    float den = __shfl(acc[r], (lane + 4) & 31, 32);
    float d = num / den * stride;
    if (n < 4) ldsDist[hi * 8 + r][n] = d;   // D row M = r + 8*hi
  }
  __syncthreads();

  // ---- assemble boxes (lanes 0..15, one anchor each) ----
  if (hi == 0 && j < NPRE) {
    int row = sel / W, col = sel % W;
    float py = ((float)row + 0.5f) * stride;
    float px = ((float)col + 0.5f) * stride;
    float d0 = ldsDist[m][0], d1 = ldsDist[m][1];
    float d2 = ldsDist[m][2], d3 = ldsDist[m][3];
    float by1 = fminf(fmaxf(py - d0, 0.f), 1024.f);
    float bx1 = fminf(fmaxf(px - d1, 0.f), 1024.f);
    float by2 = fminf(fmaxf(py + d2, 0.f), 1024.f);
    float bx2 = fminf(fmaxf(px + d3, 0.f), 1024.f);
    float* bo = boxes + ((size_t)b * NTOT + lvl * NPRE + j) * 4;
    bo[0] = bx1; bo[1] = by1; bo[2] = bx2; bo[3] = by2;  // (x1,y1,x2,y2)
  }

  // ---- sigmoid scores, class-major layout for NMS ----
  if (j < NPRE) {
    const float* cp = cls + ((size_t)(b * HW + sel)) * C_;
    const int c0 = hi ? 40 : 0;
    const int slot = lvl * NPRE + j;
    #pragma unroll 4
    for (int cc = 0; cc < 40; ++cc) {
      int c = c0 + cc;
      float s = 1.0f / (1.0f + expf(-cp[c]));
      scoresT[((size_t)b * C_ + c) * NTOT + slot] = s;
    }
  }
}

// ---------------- Kernel 4: greedy NMS per (image,class), LDS-resident -----
__global__ void k_nms(const float* __restrict__ boxes,
                      const float* __restrict__ scoresT,
                      float* __restrict__ nms_b, float* __restrict__ nms_s) {
  __shared__ float  sc[NTOT];      // 12 KB
  __shared__ float4 bxs[NTOT];     // 48 KB
  __shared__ float  redv[256];
  __shared__ int    redi[256];
  const int c = blockIdx.x, b = blockIdx.y, tid = threadIdx.x;
  const float4* bsrc = (const float4*)(boxes + (size_t)b * NTOT * 4);
  const float*  ssrc = scoresT + ((size_t)b * C_ + c) * NTOT;
  for (int i = tid; i < NTOT; i += 256) { bxs[i] = bsrc[i]; sc[i] = ssrc[i]; }
  __syncthreads();

  float* ob = nms_b + ((size_t)(b * C_ + c)) * K_ * 4;
  float* os = nms_s + ((size_t)(b * C_ + c)) * K_;
  for (int it = 0; it < K_; ++it) {
    // block argmax over sc
    float bv = NINF; int bi = 0;
    for (int i = tid; i < NTOT; i += 256) { float v = sc[i]; if (v > bv) { bv = v; bi = i; } }
    redv[tid] = bv; redi[tid] = bi;
    __syncthreads();
    for (int s = 128; s > 0; s >>= 1) {
      if (tid < s && redv[tid + s] > redv[tid]) { redv[tid] = redv[tid + s]; redi[tid] = redi[tid + s]; }
      __syncthreads();
    }
    const int   seli = redi[0];
    const float sval = redv[0];
    const float4 sb  = bxs[seli];
    if (tid == 0) {
      os[it] = sval;
      ob[it * 4 + 0] = sb.x; ob[it * 4 + 1] = sb.y;
      ob[it * 4 + 2] = sb.z; ob[it * 4 + 3] = sb.w;
    }
    // suppress IoU > 0.5 (reference formula, strict >)
    const float a0 = fmaxf(sb.z - sb.x, 0.f) * fmaxf(sb.w - sb.y, 0.f);
    for (int i = tid; i < NTOT; i += 256) {
      float4 o = bxs[i];
      float xx1 = fmaxf(sb.x, o.x), yy1 = fmaxf(sb.y, o.y);
      float xx2 = fminf(sb.z, o.z), yy2 = fminf(sb.w, o.w);
      float inter = fmaxf(xx2 - xx1, 0.f) * fmaxf(yy2 - yy1, 0.f);
      float a1 = fmaxf(o.z - o.x, 0.f) * fmaxf(o.w - o.y, 0.f);
      float iou = inter / fmaxf(a0 + a1 - inter, 1e-9f);
      if (iou > 0.5f) sc[i] = NINF;
    }
    __syncthreads();
  }
}

// ---------------- Kernel 5: global top-100 per image + assembly ------------
__global__ void k_final(const float* __restrict__ nms_b,
                        const float* __restrict__ nms_s,
                        float* __restrict__ out) {
  __shared__ float fs[C_ * K_];    // 32 KB
  __shared__ float redv[256];
  __shared__ int   redi[256];
  const int b = blockIdx.x, tid = threadIdx.x;
  const float* ssrc = nms_s + (size_t)b * C_ * K_;
  for (int i = tid; i < C_ * K_; i += 256) fs[i] = ssrc[i];
  __syncthreads();
  for (int k = 0; k < K_; ++k) {
    float bv = NINF; int bi = 0;
    for (int i = tid; i < C_ * K_; i += 256) { float v = fs[i]; if (v > bv) { bv = v; bi = i; } }
    redv[tid] = bv; redi[tid] = bi;
    __syncthreads();
    for (int s = 128; s > 0; s >>= 1) {
      if (tid < s && redv[tid + s] > redv[tid]) { redv[tid] = redv[tid + s]; redi[tid] = redi[tid + s]; }
      __syncthreads();
    }
    if (tid == 0) {
      int i = redi[0]; float v = redv[0];
      float* o = out + ((size_t)b * K_ + k) * 6;
      if (v > 0.3f) {   // finite & above threshold -> keep row
        const float* nb = nms_b + ((size_t)b * C_ * K_ + i) * 4;
        o[0] = nb[0]; o[1] = nb[1]; o[2] = nb[2]; o[3] = nb[3];
        o[4] = v; o[5] = (float)(i / K_);
      } else {
        o[0] = 0.f; o[1] = 0.f; o[2] = 0.f; o[3] = 0.f; o[4] = 0.f; o[5] = 0.f;
      }
      fs[i] = NINF;
    }
    __syncthreads();
  }
}

// ---------------------------------------------------------------------------
extern "C" void kernel_launch(void* const* d_in, const int* in_sizes, int n_in,
                              void* d_out, int out_size, void* d_ws, size_t ws_size,
                              hipStream_t stream) {
  (void)in_sizes; (void)n_in; (void)out_size; (void)ws_size;
  const float* cls0 = (const float*)d_in[0];
  const float* bp0  = (const float*)d_in[1];
  const float* cls1 = (const float*)d_in[2];
  const float* bp1  = (const float*)d_in[3];
  const float* cls2 = (const float*)d_in[4];
  const float* bp2  = (const float*)d_in[5];
  float* out = (float*)d_out;
  char* ws = (char*)d_ws;

  // workspace layout (bytes, 16B aligned)
  constexpr size_t OFF_KEY = 0;                    // 8*21504 u32   = 688128
  constexpr size_t OFF_SEL = OFF_KEY + 688128;     // 8*3*1000 int  =  96000
  constexpr size_t OFF_BOX = OFF_SEL + 96000;      // 8*3000*4 f32  = 384000
  constexpr size_t OFF_SCO = OFF_BOX + 384000;     // 8*80*3000 f32 = 7680000
  constexpr size_t OFF_NS  = OFF_SCO + 7680000;    // 8*80*100 f32  = 256000
  constexpr size_t OFF_NB  = OFF_NS + 256000;      // 8*80*100*4 f32= 1024000
  unsigned* maxkey  = (unsigned*)(ws + OFF_KEY);
  int*      sel     = (int*)     (ws + OFF_SEL);
  float*    boxes   = (float*)   (ws + OFF_BOX);
  float*    scoresT = (float*)   (ws + OFF_SCO);
  float*    nms_s   = (float*)   (ws + OFF_NS);
  float*    nms_b   = (float*)   (ws + OFF_NB);

  k_maxkey<<<dim3((B_ * 16384 + 255) / 256), 256, 0, stream>>>(cls0, maxkey, B_ * 16384);
  k_maxkey<<<dim3((B_ * 4096  + 255) / 256), 256, 0, stream>>>(cls1, maxkey + B_ * 16384, B_ * 4096);
  k_maxkey<<<dim3((B_ * 1024  + 255) / 256), 256, 0, stream>>>(cls2, maxkey + B_ * 16384 + B_ * 4096, B_ * 1024);
  k_select<<<dim3(3, B_), 256, 0, stream>>>(maxkey, sel);
  k_decode<<<dim3(63, 3, B_), 32, 0, stream>>>(cls0, bp0, cls1, bp1, cls2, bp2,
                                               sel, boxes, scoresT);
  k_nms<<<dim3(C_, B_), 256, 0, stream>>>(boxes, scoresT, nms_b, nms_s);
  k_final<<<dim3(B_), 256, 0, stream>>>(nms_b, nms_s, out);
}
